// MultiHeadAttention_77189152244223
// MI455X (gfx1250) — matrix operand order repaired and checked by reading, not compiled, not run on hardware
//
#include <hip/hip_runtime.h>
#include <hip/hip_bf16.h>

typedef __attribute__((ext_vector_type(16))) __bf16 v16bf;
typedef __attribute__((ext_vector_type(8)))  __bf16 v8bf;
typedef __attribute__((ext_vector_type(4)))  __bf16 v4bf;
typedef __attribute__((ext_vector_type(8)))  float  v8f;
typedef __attribute__((ext_vector_type(4)))  float  v4f;

#define HEADS   16
#define DMODEL  1024
#define DK      64
#define BATCH   4
#define SEQ     2048
#define MROWS   (BATCH * SEQ)   // 8192

__device__ __forceinline__ v8f wmma_bf16(v16bf a, v16bf b, v8f c) {
    return __builtin_amdgcn_wmma_f32_16x16x32_bf16(false, a, false, b, (short)0, c, false, false);
}

__device__ __forceinline__ v16bf join16(v8bf lo, v8bf hi) {
    return __builtin_shufflevector(lo, hi, 0, 1, 2, 3, 4, 5, 6, 7,
                                           8, 9, 10, 11, 12, 13, 14, 15);
}

// A-fragment (16x32): p -> 32 contiguous bf16 (k=0..31) for this lane's row.
__device__ __forceinline__ v16bf load_afrag(const __bf16* p, int hf) {
    const v8bf* q = (const v8bf*)p;
    return join16(q[2 * hf], q[2 * hf + 1]);
}

// B-fragment (32x16): p -> 16 contiguous bf16, k = (caller adds hf*16) + e
__device__ __forceinline__ v16bf load_bfrag(const __bf16* p) {
    const v8bf* q = (const v8bf*)p;
    return join16(q[0], q[1]);
}

// CDNA5 async global->LDS 128-bit copy (ASYNCcnt-tracked, no VGPR round trip).
__device__ __forceinline__ void async_copy16(void* lds_dst, const void* gsrc) {
    unsigned ldsoff = (unsigned)(unsigned long long)lds_dst;
    asm volatile("global_load_async_to_lds_b128 %0, %1, off"
                 :: "v"(ldsoff), "v"(gsrc)
                 : "memory");
}
__device__ __forceinline__ void async_wait0() {
    asm volatile("s_wait_asynccnt 0x0" ::: "memory");
}

// ---------------------------------------------------------------------------
// Prep 1: f32 -> bf16 stream convert (8 elements / thread)
// ---------------------------------------------------------------------------
__global__ __launch_bounds__(256) void cvt_kernel(
    const float* __restrict__ src, __bf16* __restrict__ dst)
{
    size_t base = ((size_t)blockIdx.x * 256 + threadIdx.x) * 8;
    v4f a = *(const v4f*)&src[base];
    v4f b = *(const v4f*)&src[base + 4];
    v8bf o;
    #pragma unroll
    for (int j = 0; j < 4; ++j) { o[j] = (__bf16)a[j]; o[4 + j] = (__bf16)b[j]; }
    *(v8bf*)&dst[base] = o;
}

// ---------------------------------------------------------------------------
// Prep 2: W[1024x1024] f32 -> WT[n][k] bf16 (LDS-tiled transpose, once per W)
// ---------------------------------------------------------------------------
__global__ __launch_bounds__(256) void wtrans_kernel(
    const float* __restrict__ W, __bf16* __restrict__ WT)
{
    __shared__ alignas(16) __bf16 Ls[64][72];
    const int tid = threadIdx.x;
    const int n0 = blockIdx.x * 64, k0 = blockIdx.y * 64;
    #pragma unroll
    for (int i = 0; i < 4; ++i) {
        int idx = tid + i * 256;
        int r = idx >> 4, cs = (idx & 15) * 4;
        v4f wv = *(const v4f*)&W[(size_t)(k0 + r) * DMODEL + n0 + cs];
        v4bf wb;
        #pragma unroll
        for (int j = 0; j < 4; ++j) wb[j] = (__bf16)wv[j];
        *(v4bf*)&Ls[r][cs] = wb;
    }
    __syncthreads();
    #pragma unroll
    for (int i = 0; i < 4; ++i) {
        int idx = tid + i * 256;
        int n = idx >> 4, ks = (idx & 15) * 4;
        v4bf ob;
        #pragma unroll
        for (int j = 0; j < 4; ++j) ob[j] = Ls[ks + j][n];
        *(v4bf*)&WT[(size_t)(n0 + n) * DMODEL + k0 + ks] = ob;
    }
}

// ---------------------------------------------------------------------------
// Projection GEMM (bf16): 128x64 block tile, double-buffered async-LDS
// staging pipelined against 8 WMMAs per wave per K-step.
// mode 0: dst layout [B,H,S,DK]   (Q, K)
// mode 1: dst layout [B,H,DK,S]   (V transposed)
// ---------------------------------------------------------------------------
__global__ __launch_bounds__(128) void gemm_proj(
    const __bf16* __restrict__ Xb, const __bf16* __restrict__ WT,
    const float* __restrict__ bias, __bf16* __restrict__ dst, int mode)
{
    __shared__ alignas(16) __bf16 As[2][128][40];   // 2 x (128 x 32 +pad)
    __shared__ alignas(16) __bf16 BsT[2][64][40];   // 2 x ([n][k] 64 x 32 +pad)

    const int tid  = threadIdx.x;
    const int wave = tid >> 5, lane = tid & 31;
    const int hf   = lane >> 4, l16 = lane & 15;
    const int n0   = blockIdx.x * 64;
    const int m0   = blockIdx.y * 128;

    v8f acc0[4] = {};
    v8f acc1[4] = {};

    auto stage = [&](int buf, int kb) {
        #pragma unroll
        for (int i = 0; i < 4; ++i) {
            int idx = tid + i * 128;
            int r = idx >> 2, seg = (idx & 3) * 8;
            async_copy16(&As[buf][r][seg], &Xb[(size_t)(m0 + r) * DMODEL + kb + seg]);
        }
        #pragma unroll
        for (int i = 0; i < 2; ++i) {
            int idx = tid + i * 128;
            int r = idx >> 2, seg = (idx & 3) * 8;
            async_copy16(&BsT[buf][r][seg], &WT[(size_t)(n0 + r) * DMODEL + kb + seg]);
        }
    };

    stage(0, 0);
    const int NIT = DMODEL / 32;
    for (int it = 0; it < NIT; ++it) {
        int cur = it & 1;
        async_wait0();          // my copies into cur done
        __syncthreads();        // everyone's copies into cur done
        if (it + 1 < NIT)
            stage(cur ^ 1, (it + 1) * 32);   // DMA next tile under the WMMAs

        v16bf a0 = load_afrag(&As[cur][wave * 32 + l16][0], hf);
        v16bf a1 = load_afrag(&As[cur][wave * 32 + 16 + l16][0], hf);
        #pragma unroll
        for (int c4 = 0; c4 < 4; ++c4) {
            v16bf bf = load_bfrag(&BsT[cur][c4 * 16 + l16][hf * 16]);
            acc0[c4] = wmma_bf16(a0, bf, acc0[c4]);
            acc1[c4] = wmma_bf16(a1, bf, acc1[c4]);
        }
        // no trailing barrier: cur is re-staged only after the next
        // wait_asynccnt + barrier, and all ds_loads complete before their
        // consuming WMMAs issue.
    }

    #pragma unroll
    for (int g = 0; g < 2; ++g) {
        v8f* acc = g ? acc1 : acc0;
        #pragma unroll
        for (int c4 = 0; c4 < 4; ++c4) {
            #pragma unroll
            for (int r = 0; r < 8; ++r) {
                int mrow = m0 + wave * 32 + g * 16 + r + hf * 8;
                int ncol = n0 + c4 * 16 + l16;
                float v  = acc[c4][r] + bias[ncol];
                int bb = mrow / SEQ, ss = mrow % SEQ;
                int hh = ncol / DK,  dd = ncol % DK;
                size_t idx = (mode == 1)
                    ? (((size_t)(bb * HEADS + hh) * DK + dd) * SEQ + ss)
                    : (((size_t)(bb * HEADS + hh) * SEQ + ss) * DK + dd);
                dst[idx] = (__bf16)v;
            }
        }
    }
}

// ---------------------------------------------------------------------------
// Attention: per wave, one 16-query tile of one (b,h).
// ---------------------------------------------------------------------------
__global__ __launch_bounds__(128) void attn_kernel(
    const __bf16* __restrict__ Qp, const __bf16* __restrict__ Kp,
    const __bf16* __restrict__ VpT, const int* __restrict__ mask,
    float* __restrict__ attn, __bf16* __restrict__ Cw)
{
    __shared__ alignas(16) float probs[4][16][36];

    const int tid  = threadIdx.x;
    const int wave = tid >> 5, lane = tid & 31;
    const int hf   = lane >> 4, l16 = lane & 15;
    const int b    = blockIdx.z, h = blockIdx.y;
    const int q0   = (blockIdx.x * 4 + wave) * 16;

    const size_t  bh      = (size_t)b * HEADS + h;
    const __bf16* Qbh     = Qp  + bh * SEQ * DK;
    const __bf16* Kbh     = Kp  + bh * SEQ * DK;
    const __bf16* Vbh     = VpT + bh * DK * SEQ;
    float*        attn_bh = attn + bh * (size_t)SEQ * SEQ;

    const __bf16* qrow = Qbh + (size_t)(q0 + l16) * DK;
    v16bf aq0 = load_afrag(qrow, hf);
    v16bf aq1 = load_afrag(qrow + 32, hf);

    float mrow[8];
    #pragma unroll
    for (int r = 0; r < 8; ++r) mrow[r] = -1e30f;

    // ---- pass 1: scores ----
    for (int kt = 0; kt < SEQ / 16; ++kt) {
        int key = kt * 16 + l16;
        const __bf16* krow = Kbh + (size_t)key * DK;
        v16bf b0 = load_bfrag(krow + hf * 16);
        v16bf b1 = load_bfrag(krow + 32 + hf * 16);
        v8f sc = {};
        sc = wmma_bf16(aq0, b0, sc);
        sc = wmma_bf16(aq1, b1, sc);
        #pragma unroll
        for (int r = 0; r < 8; ++r) {
            int q = q0 + r + hf * 8;
            float s = sc[r] * 0.125f;
            int mk = mask[((size_t)b * SEQ + q) * SEQ + key];
            s = (mk == 0) ? -1e9f : s;
            attn_bh[(size_t)q * SEQ + key] = s;
            mrow[r] = fmaxf(mrow[r], s);
        }
    }
    #pragma unroll
    for (int r = 0; r < 8; ++r)
        #pragma unroll
        for (int off = 1; off < 16; off <<= 1)
            mrow[r] = fmaxf(mrow[r], __shfl_xor(mrow[r], off, 32));

    // ---- pass 2: exp + ctx ----
    float ssum[8] = {0.f, 0.f, 0.f, 0.f, 0.f, 0.f, 0.f, 0.f};
    v8f ctx[4] = {};
    for (int kt2 = 0; kt2 < SEQ / 32; ++kt2) {
        #pragma unroll
        for (int sub = 0; sub < 2; ++sub) {
            int kt = kt2 * 2 + sub;
            #pragma unroll
            for (int r = 0; r < 8; ++r) {
                int q = q0 + r + hf * 8;
                int key = kt * 16 + l16;
                float s = attn_bh[(size_t)q * SEQ + key];
                float p = __expf(s - mrow[r]);
                ssum[r] += p;
                probs[wave][r + hf * 8][sub * 16 + l16] = p;
            }
        }
        const float* pr = &probs[wave][l16][0];
        v4f f0 = *(const v4f*)(pr + hf * 16);
        v4f f1 = *(const v4f*)(pr + hf * 16 + 4);
        v4f f2 = *(const v4f*)(pr + hf * 16 + 8);
        v4f f3 = *(const v4f*)(pr + hf * 16 + 12);
        v16bf ap;
        #pragma unroll
        for (int j = 0; j < 4; ++j) {
            ap[j]      = (__bf16)f0[j];
            ap[4 + j]  = (__bf16)f1[j];
            ap[8 + j]  = (__bf16)f2[j];
            ap[12 + j] = (__bf16)f3[j];
        }
        #pragma unroll
        for (int c4 = 0; c4 < 4; ++c4) {
            const __bf16* vrow = Vbh + (size_t)(c4 * 16 + l16) * SEQ + kt2 * 32 + hf * 16;
            v16bf bv = load_bfrag(vrow);
            ctx[c4] = wmma_bf16(ap, bv, ctx[c4]);
        }
    }
    #pragma unroll
    for (int r = 0; r < 8; ++r) {
        #pragma unroll
        for (int off = 1; off < 16; off <<= 1)
            ssum[r] += __shfl_xor(ssum[r], off, 32);
        ssum[r] = 1.0f / ssum[r];
    }
    #pragma unroll
    for (int c4 = 0; c4 < 4; ++c4)
        #pragma unroll
        for (int r = 0; r < 8; ++r) {
            int q = q0 + r + hf * 8;
            int d = c4 * 16 + l16;
            Cw[((size_t)b * SEQ + q) * DMODEL + h * DK + d] =
                (__bf16)(ctx[c4][r] * ssum[r]);
        }
    // ---- pass 3: normalize attn in place ----
    for (int kt = 0; kt < SEQ / 16; ++kt) {
        #pragma unroll
        for (int r = 0; r < 8; ++r) {
            int q = q0 + r + hf * 8;
            int key = kt * 16 + l16;
            size_t ix = (size_t)q * SEQ + key;
            attn_bh[ix] = __expf(attn_bh[ix] - mrow[r]) * ssum[r];
        }
    }
}

// ---------------------------------------------------------------------------
// Output projection: out = Cw * WoT^T + bo (f32 out), same pipelined tiling.
// ---------------------------------------------------------------------------
__global__ __launch_bounds__(128) void gemm_out(
    const __bf16* __restrict__ Cw, const __bf16* __restrict__ WoT,
    const float* __restrict__ bo, float* __restrict__ out)
{
    __shared__ alignas(16) __bf16 As[2][128][40];
    __shared__ alignas(16) __bf16 BsT[2][64][40];

    const int tid  = threadIdx.x;
    const int wave = tid >> 5, lane = tid & 31;
    const int hf   = lane >> 4, l16 = lane & 15;
    const int n0   = blockIdx.x * 64;
    const int m0   = blockIdx.y * 128;

    v8f acc0[4] = {};
    v8f acc1[4] = {};

    auto stage = [&](int buf, int kb) {
        #pragma unroll
        for (int i = 0; i < 4; ++i) {
            int idx = tid + i * 128;
            int r = idx >> 2, seg = (idx & 3) * 8;
            async_copy16(&As[buf][r][seg], &Cw[(size_t)(m0 + r) * DMODEL + kb + seg]);
        }
        #pragma unroll
        for (int i = 0; i < 2; ++i) {
            int idx = tid + i * 128;
            int r = idx >> 2, seg = (idx & 3) * 8;
            async_copy16(&BsT[buf][r][seg], &WoT[(size_t)(n0 + r) * DMODEL + kb + seg]);
        }
    };

    stage(0, 0);
    const int NIT = DMODEL / 32;
    for (int it = 0; it < NIT; ++it) {
        int cur = it & 1;
        async_wait0();
        __syncthreads();
        if (it + 1 < NIT)
            stage(cur ^ 1, (it + 1) * 32);

        v16bf a0 = load_afrag(&As[cur][wave * 32 + l16][0], hf);
        v16bf a1 = load_afrag(&As[cur][wave * 32 + 16 + l16][0], hf);
        #pragma unroll
        for (int c4 = 0; c4 < 4; ++c4) {
            v16bf bf = load_bfrag(&BsT[cur][c4 * 16 + l16][hf * 16]);
            acc0[c4] = wmma_bf16(a0, bf, acc0[c4]);
            acc1[c4] = wmma_bf16(a1, bf, acc1[c4]);
        }
    }

    #pragma unroll
    for (int g = 0; g < 2; ++g) {
        v8f* acc = g ? acc1 : acc0;
        #pragma unroll
        for (int c4 = 0; c4 < 4; ++c4)
            #pragma unroll
            for (int r = 0; r < 8; ++r) {
                int mrow = m0 + wave * 32 + g * 16 + r + hf * 8;
                int ncol = n0 + c4 * 16 + l16;
                out[(size_t)mrow * DMODEL + ncol] = acc[c4][r] + bo[ncol];
            }
    }
}

// ---------------------------------------------------------------------------
extern "C" void kernel_launch(void* const* d_in, const int* in_sizes, int n_in,
                              void* d_out, int out_size, void* d_ws, size_t ws_size,
                              hipStream_t stream)
{
    const float* q    = (const float*)d_in[0];
    const float* k    = (const float*)d_in[1];
    const float* v    = (const float*)d_in[2];
    const int*   mask = (const int*)  d_in[3];
    const float* Wq   = (const float*)d_in[4];
    const float* bq   = (const float*)d_in[5];
    const float* Wk   = (const float*)d_in[6];
    const float* bk   = (const float*)d_in[7];
    const float* Wv   = (const float*)d_in[8];
    const float* bv   = (const float*)d_in[9];
    const float* Wo   = (const float*)d_in[10];
    const float* bo   = (const float*)d_in[11];

    const size_t NX = (size_t)MROWS * DMODEL;   // 8M elems
    const size_t NW = (size_t)DMODEL * DMODEL;  // 1M elems

    __bf16* Xq  = (__bf16*)d_ws;
    __bf16* Xk  = Xq  + NX;
    __bf16* Xv  = Xk  + NX;
    __bf16* WTq = Xv  + NX;
    __bf16* WTk = WTq + NW;
    __bf16* WTv = WTk + NW;
    __bf16* WTo = WTv + NW;
    __bf16* Qp  = WTo + NW;                     // [B,H,S,DK]
    __bf16* Kp  = Qp  + NX;                     // [B,H,S,DK]
    __bf16* VpT = Kp  + NX;                     // [B,H,DK,S]
    __bf16* Cw  = VpT + NX;                     // [B,S,DMODEL]

    float* out  = (float*)d_out;                // [B,S,DMODEL]
    float* attn = out + NX;                     // [B,H,S,S]

    dim3 gcvt((unsigned)(NX / (256 * 8)));
    cvt_kernel<<<gcvt, 256, 0, stream>>>(q, Xq);
    cvt_kernel<<<gcvt, 256, 0, stream>>>(k, Xk);
    cvt_kernel<<<gcvt, 256, 0, stream>>>(v, Xv);
    dim3 gwt(DMODEL / 64, DMODEL / 64);
    wtrans_kernel<<<gwt, 256, 0, stream>>>(Wq, WTq);
    wtrans_kernel<<<gwt, 256, 0, stream>>>(Wk, WTk);
    wtrans_kernel<<<gwt, 256, 0, stream>>>(Wv, WTv);
    wtrans_kernel<<<gwt, 256, 0, stream>>>(Wo, WTo);

    dim3 gproj(DMODEL / 64, MROWS / 128, 1);    // (16, 64)
    gemm_proj<<<gproj, 128, 0, stream>>>(Xq, WTq, bq, Qp, 0);
    gemm_proj<<<gproj, 128, 0, stream>>>(Xk, WTk, bk, Kp, 0);
    gemm_proj<<<gproj, 128, 0, stream>>>(Xv, WTv, bv, VpT, 1);

    dim3 gattn(SEQ / 64, HEADS, BATCH);         // (32, 16, 4)
    attn_kernel<<<gattn, 128, 0, stream>>>(Qp, Kp, VpT, mask, attn, Cw);

    gemm_out<<<gproj, 128, 0, stream>>>(Cw, WTo, bo, out);
}
